// ProposedEnergyModel_41360535060872
// MI455X (gfx1250) — matrix operand order repaired and checked by reading, not compiled, not run on hardware
//
#include <hip/hip_runtime.h>
#include <hip/hip_bf16.h>

// Problem constants (match reference)
#define NUM_GRAPHS 4096
#define MODEL_DIMS 512

typedef __attribute__((ext_vector_type(16))) __bf16 v16bf;
typedef __attribute__((ext_vector_type(8)))  float  v8f;

// Types matching __builtin_amdgcn_global_load_async_to_lds_b128's parameters:
// (int4 addrspace(1)*, int4 addrspace(3)*, imm int, imm int)
typedef int v4i __attribute__((vector_size(16)));
typedef __attribute__((address_space(1))) v4i as1_v4i;
typedef __attribute__((address_space(3))) v4i as3_v4i;
typedef __attribute__((address_space(3))) void as3_void;

union FragAB {
    v16bf v;
    uint4 q[2];
};

__device__ __forceinline__ float gelu_tanh(float x) {
    // jax.nn.gelu default (approximate=True): 0.5x(1+tanh(sqrt(2/pi)(x+0.044715x^3)))
    float x3 = x * x * x;
    return 0.5f * x * (1.0f + tanhf(0.7978845608028654f * (x + 0.044715f * x3)));
}

// ---------------------------------------------------------------------------
// Async global->LDS 16B copy (gfx1250 GLOBAL_LOAD_ASYNC_TO_LDS_B128, ASYNCcnt)
// ---------------------------------------------------------------------------
__device__ __forceinline__ void async_copy_16B(const void* g, void* l) {
#if __has_builtin(__builtin_amdgcn_global_load_async_to_lds_b128)
    __builtin_amdgcn_global_load_async_to_lds_b128(
        (as1_v4i*)(unsigned long long)g,   // global src (AS1)
        (as3_v4i*)l,                       // LDS dst (AS3, addrspacecast)
        0, 0);
#else
    unsigned lofs = (unsigned)(unsigned long long)(as3_void*)l;
    asm volatile("global_load_async_to_lds_b128 %0, %1, off"
                 :: "v"(lofs), "v"((unsigned long long)g)
                 : "memory");
#endif
}

__device__ __forceinline__ void async_wait_all() {
#if __has_builtin(__builtin_amdgcn_s_wait_asynccnt)
    __builtin_amdgcn_s_wait_asynccnt(0);
#else
    asm volatile("s_wait_asynccnt 0x0" ::: "memory");
#endif
}

// ---------------------------------------------------------------------------
// Kernel 1: energy[g] = b2[0]  (seed output with final bias)
// ---------------------------------------------------------------------------
__global__ void init_energy_kernel(const float* __restrict__ b2,
                                   float* __restrict__ energy) {
    int g = blockIdx.x * blockDim.x + threadIdx.x;
    if (g < NUM_GRAPHS) energy[g] = b2[0];
}

// ---------------------------------------------------------------------------
// Kernel 2: fused per-atom feature + segment sum (batch is sorted).
// Block = 512 threads (one per model dim), 128 atoms per block, register
// accumulation per graph-run, atomic flush only at run boundaries.
// ---------------------------------------------------------------------------
__global__ __launch_bounds__(512) void segsum_kernel(
    const int*   __restrict__ atomic_numbers,
    const float* __restrict__ pos,
    const int*   __restrict__ batch,
    const float* __restrict__ embed,
    const float* __restrict__ w_pos,
    float* __restrict__ summed,
    float* __restrict__ counts,
    int num_atoms)
{
    const int d    = threadIdx.x;              // 0..511, one dim per thread
    const int base = blockIdx.x * 128;

    const float wp0 = w_pos[d];
    const float wp1 = w_pos[MODEL_DIMS + d];
    const float wp2 = w_pos[2 * MODEL_DIMS + d];

    float acc = 0.0f;
    int   cur = -1;
    int   runlen = 0;

    for (int ii = 0; ii < 128; ++ii) {
        int i = base + ii;
        if (i >= num_atoms) break;
        int g = batch[i];                      // uniform across block
        if (g != cur) {
            if (cur >= 0) {
                atomicAdd(&summed[(size_t)cur * MODEL_DIMS + d], acc);
                if (d == 0) atomicAdd(&counts[cur], (float)runlen);
            }
            acc = 0.0f; runlen = 0; cur = g;
        }
        int an = atomic_numbers[i];
        float p0 = pos[3 * i + 0];
        float p1 = pos[3 * i + 1];
        float p2 = pos[3 * i + 2];
        acc += embed[(size_t)an * MODEL_DIMS + d] + p0 * wp0 + p1 * wp1 + p2 * wp2;
        runlen++;
    }
    if (cur >= 0) {
        atomicAdd(&summed[(size_t)cur * MODEL_DIMS + d], acc);
        if (d == 0) atomicAdd(&counts[cur], (float)runlen);
    }
}

// ---------------------------------------------------------------------------
// Kernel 3: reduced = summed / max(counts,1), cast to bf16 for WMMA A-matrix
// ---------------------------------------------------------------------------
__global__ void finalize_reduced_kernel(const float* __restrict__ summed,
                                        const float* __restrict__ counts,
                                        __bf16* __restrict__ redb) {
    int idx = blockIdx.x * blockDim.x + threadIdx.x;
    if (idx >= NUM_GRAPHS * MODEL_DIMS) return;
    int g = idx >> 9;                          // / MODEL_DIMS
    float c = counts[g];
    c = c > 1.0f ? c : 1.0f;
    redb[idx] = (__bf16)(summed[idx] / c);
}

// ---------------------------------------------------------------------------
// Kernel 4: cast w1 to bf16 (B-matrix)
// ---------------------------------------------------------------------------
__global__ void convert_w1_kernel(const float* __restrict__ w1,
                                  __bf16* __restrict__ w1b) {
    int idx = blockIdx.x * blockDim.x + threadIdx.x;
    if (idx < MODEL_DIMS * MODEL_DIMS) w1b[idx] = (__bf16)w1[idx];
}

// ---------------------------------------------------------------------------
// Kernel 5: WMMA GEMM  h = gelu(reduced @ w1 + b1), fused epilogue
//           energy[m] += h[m][n] * w2[n]   (atomic, seeded with b2)
//
// Block = 8 waves, owns one N-tile (16 cols of w1) and 16 M-tiles (2/wave).
// The 512x16 bf16 B panel (16 KB) is staged into LDS ONCE per block via
// GLOBAL_LOAD_ASYNC_TO_LDS_B128 (ASYNCcnt), rows padded to 48 B to spread
// LDS banks; waves then feed V_WMMA_F32_16X16X32_BF16 with A fragments from
// global (b128) and B fragments from LDS (16 bf16 loads per lane).
//
// Fragment layouts per CDNA5 ISA:
//  A (16x32 bf16): lane -> M (lane&15); half=lane>>4; two contiguous 16B runs
//    per lane: [k0+8h .. +7] and [k0+16+8h .. +7].
//  B (32x16 bf16): lane -> N (lane&15); 16 K values per lane, same k order as A.
//  C/D (16x16 f32): lane -> N (lane&15); VGPR r -> M = r + 8*(lane>>4).
// ---------------------------------------------------------------------------
#define B_ROW_BYTES 48   // 32B of data + 16B pad (16B-aligned for b128)

__global__ __launch_bounds__(256) void wmma_mlp_kernel(
    const __bf16* __restrict__ redb,   // [4096][512] bf16 row-major
    const __bf16* __restrict__ w1b,    // [512][512]  bf16 row-major
    const float*  __restrict__ b1,     // [512]
    const float*  __restrict__ w2,     // [512]
    float* __restrict__ energy)        // [4096], pre-seeded with b2
{
    __shared__ __align__(16) unsigned char ldsB[MODEL_DIMS * B_ROW_BYTES]; // 24 KB

    const int tid    = threadIdx.x;
    const int lane   = tid & 31;
    const int warp   = tid >> 5;
    const int tn     = blockIdx.x & 31;        // N tile 0..31
    const int mchunk = blockIdx.x >> 5;        // 0..15
    const int n0     = tn * 16;

    // ---- Stage B panel w1b[:, n0:n0+16] into LDS (1024 x 16B async copies)
    for (int c = tid; c < 2 * MODEL_DIMS; c += 256) {
        int k = c >> 1, h = c & 1;
        const __bf16* g = w1b + (size_t)k * MODEL_DIMS + n0 + h * 8;
        void* l = ldsB + k * B_ROW_BYTES + h * 16;
        async_copy_16B(g, l);
    }
    async_wait_all();
    __syncthreads();

    const int half = lane >> 4;
    const int nl   = lane & 15;
    const int n    = n0 + nl;
    const int tm0  = mchunk * 16 + warp * 2;   // first of this wave's 2 M-tiles

    const __bf16* arow0 = redb + (size_t)(tm0 * 16 + nl) * MODEL_DIMS;
    const __bf16* arow1 = arow0 + (size_t)16 * MODEL_DIMS;
    const unsigned char* bl = ldsB + nl * 2;  // B column n0+nl

    v8f c0 = {0.f, 0.f, 0.f, 0.f, 0.f, 0.f, 0.f, 0.f};
    v8f c1 = {0.f, 0.f, 0.f, 0.f, 0.f, 0.f, 0.f, 0.f};

    #pragma unroll 4
    for (int k0 = 0; k0 < MODEL_DIMS; k0 += 32) {
        FragAB b;
        #pragma unroll
        for (int s = 0; s < 16; ++s) {
            int kk = k0 + (s < 8 ? 8 * half + s : 16 + 8 * half + (s - 8));
            b.v[s] = *(const __bf16*)(bl + (size_t)kk * B_ROW_BYTES);
        }
        FragAB a0, a1;
        a0.q[0] = *(const uint4*)(arow0 + k0 + 8 * half);
        a0.q[1] = *(const uint4*)(arow0 + k0 + 16 + 8 * half);
        a1.q[0] = *(const uint4*)(arow1 + k0 + 8 * half);
        a1.q[1] = *(const uint4*)(arow1 + k0 + 16 + 8 * half);
        c0 = __builtin_amdgcn_wmma_f32_16x16x32_bf16(
                 false, a0.v, false, b.v, (short)0, c0, false, false);
        c1 = __builtin_amdgcn_wmma_f32_16x16x32_bf16(
                 false, a1.v, false, b.v, (short)0, c1, false, false);
    }

    const float bias = b1[n];
    const float w2n  = w2[n];
    #pragma unroll
    for (int r = 0; r < 8; ++r) {
        int   mr0 = tm0 * 16 + r + 8 * half;
        float h0  = gelu_tanh(c0[r] + bias);
        atomicAdd(&energy[mr0], h0 * w2n);
        float h1  = gelu_tanh(c1[r] + bias);
        atomicAdd(&energy[mr0 + 16], h1 * w2n);
    }
}

// ---------------------------------------------------------------------------
// Launch
// ---------------------------------------------------------------------------
extern "C" void kernel_launch(void* const* d_in, const int* in_sizes, int n_in,
                              void* d_out, int out_size, void* d_ws, size_t ws_size,
                              hipStream_t stream) {
    const int*   atomic_numbers = (const int*)  d_in[0];
    const float* pos            = (const float*)d_in[1];
    const int*   batch          = (const int*)  d_in[2];
    const float* embed          = (const float*)d_in[3];
    const float* w_pos          = (const float*)d_in[4];
    const float* w1             = (const float*)d_in[5];
    const float* b1             = (const float*)d_in[6];
    const float* w2             = (const float*)d_in[7];
    const float* b2             = (const float*)d_in[8];
    float* energy = (float*)d_out;

    const int num_atoms = in_sizes[0];

    // Workspace layout
    float*  summed = (float*)d_ws;                                  // 4096*512 f32
    float*  counts = summed + (size_t)NUM_GRAPHS * MODEL_DIMS;      // 4096 f32
    __bf16* redb   = (__bf16*)(counts + NUM_GRAPHS);                // 4096*512 bf16
    __bf16* w1b    = redb + (size_t)NUM_GRAPHS * MODEL_DIMS;        // 512*512 bf16

    // Zero accumulators (graph-capturable memset)
    (void)hipMemsetAsync(d_ws, 0,
                         ((size_t)NUM_GRAPHS * MODEL_DIMS + NUM_GRAPHS) * sizeof(float),
                         stream);

    init_energy_kernel<<<NUM_GRAPHS / 256, 256, 0, stream>>>(b2, energy);

    int seg_blocks = (num_atoms + 127) / 128;
    segsum_kernel<<<seg_blocks, 512, 0, stream>>>(
        atomic_numbers, pos, batch, embed, w_pos, summed, counts, num_atoms);

    finalize_reduced_kernel<<<(NUM_GRAPHS * MODEL_DIMS) / 256, 256, 0, stream>>>(
        summed, counts, redb);

    convert_w1_kernel<<<(MODEL_DIMS * MODEL_DIMS) / 256, 256, 0, stream>>>(w1, w1b);

    // 32 N-tiles x 16 M-chunks = 512 blocks; 8 waves x 2 M-tiles per block
    wmma_mlp_kernel<<<512, 256, 0, stream>>>(redb, w1b, b1, w2, energy);
}
